// CustomLSTM_53919019434481
// MI455X (gfx1250) — matrix operand order repaired and checked by reading, not compiled, not run on hardware
//
#include <hip/hip_runtime.h>

#define IN_F 118
#define H1   256
#define H2   128
#define SEQ  512
#define K1R  374
#define K2P  384
#define NT1  64       // 1024 / 16
#define NT2  32       // 512 / 16
#define KT   12       // 384 / 32
#define CA_STRIDE 392 // bf16 row stride, padded to kill LDS bank conflicts
#define XSTRIDE   120 // f32 x-stage row stride
#define EPSL 1e-5f

typedef __attribute__((ext_vector_type(16))) __bf16          v16bf;
typedef __attribute__((ext_vector_type(16))) unsigned short  v16u;
typedef __attribute__((ext_vector_type(8)))  float           v8f;

#if __has_builtin(__builtin_amdgcn_global_load_async_to_lds_b32) && \
    __has_builtin(__builtin_amdgcn_s_wait_asynccnt)
#define USE_ASYNC 1
#else
#define USE_ASYNC 0
#endif

static __device__ __forceinline__ unsigned short f2bf(float x) {
  unsigned u = __builtin_bit_cast(unsigned, x);
  unsigned r = u + 0x7FFFu + ((u >> 16) & 1u);   // round-to-nearest-even
  return (unsigned short)(r >> 16);
}
static __device__ __forceinline__ float sigm(float x) {
  return 1.0f / (1.0f + __expf(-x));
}
// branch-free tanh: tanh(|x|) = (1 - e^-2|x|)/(1 + e^-2|x|), restore sign
static __device__ __forceinline__ float tanh_fast(float x) {
  float e = __expf(-2.0f * __builtin_fabsf(x));
  float r = (1.0f - e) / (1.0f + e);
  return __builtin_copysignf(r, x);
}

#if USE_ASYNC
static __device__ __forceinline__ void async_cp_b32(const float* g, const float* l) {
  __builtin_amdgcn_global_load_async_to_lds_b32(
      (__attribute__((address_space(1))) int*)(unsigned long long)g,
      (__attribute__((address_space(3))) int*)(unsigned)(unsigned long long)l,
      0, 0);
}
#endif

// --- A fragment from LDS: contiguous K layout matching packed B (M = lane%16;
// lanes 0-15 hold K 0-15, lanes 16-31 hold K 16-31, in pairs). 32B
// contiguous -> compiler merges into ds_load_b128. ---
static __device__ __forceinline__ v16bf load_a_frag(const unsigned short* sm,
                                                    int m, int kt, int hi) {
  const unsigned short* row = sm + m * CA_STRIDE + kt * 32 + (hi ? 16 : 0);
  v16u a;
#pragma unroll
  for (int r = 0; r < 8; ++r) {
    const int kb = 2 * r;
    unsigned pair = *(const unsigned*)(row + kb);
    a[2 * r]     = (unsigned short)(pair & 0xFFFFu);
    a[2 * r + 1] = (unsigned short)(pair >> 16);
  }
  return __builtin_bit_cast(v16bf, a);
}

// --- B fragment: pre-packed so lane l reads 32 contiguous bytes -> 2x b128 ---
static __device__ __forceinline__ v16bf load_b_frag(const unsigned short* wp,
                                                    int tile, int lane) {
  const v16u* p = (const v16u*)(wp + (size_t)tile * 512 + lane * 16);
  return __builtin_bit_cast(v16bf, *p);
}

// ===================== weight packing (run once) =====================
__global__ void pack_w1(const float* __restrict__ Wf, const float* __restrict__ Wi,
                        const float* __restrict__ Wg, const float* __restrict__ Wo,
                        unsigned short* __restrict__ out) {
  int idx = blockIdx.x * 256 + threadIdx.x;
  if (idx >= NT1 * KT * 512) return;
  int tile = idx >> 9, e = idx & 511;
  int ntile = tile / KT, kt = tile % KT;
  int l = e >> 4, h = e & 15;
  int nl = l & 15, hi = l >> 4;
  int r = h >> 1, p = h & 1;
  int kl = 2 * r + p + (hi ? 16 : 0);           // B layout: lanes 0-15 K0-15, 16-31 K16-31
  int n = ntile * 16 + nl;
  int k = kt * 32 + kl;
  int gate = n >> 8, row = n & 255;
  const float* W = (gate == 0) ? Wf : (gate == 1) ? Wi : (gate == 2) ? Wg : Wo;
  float v = (k < K1R) ? W[row * K1R + k] : 0.0f;
  out[idx] = f2bf(v);
}

__global__ void pack_w2(const float* __restrict__ Wf, const float* __restrict__ Wi,
                        const float* __restrict__ Wg, const float* __restrict__ Wo,
                        unsigned short* __restrict__ out) {
  int idx = blockIdx.x * 256 + threadIdx.x;
  if (idx >= NT2 * KT * 512) return;
  int tile = idx >> 9, e = idx & 511;
  int ntile = tile / KT, kt = tile % KT;
  int l = e >> 4, h = e & 15;
  int nl = l & 15, hi = l >> 4;
  int r = h >> 1, p = h & 1;
  int kl = 2 * r + p + (hi ? 16 : 0);
  int n = ntile * 16 + nl;
  int k = kt * 32 + kl;
  int gate = n >> 7, row = n & 127;
  const float* W = (gate == 0) ? Wf : (gate == 1) ? Wi : (gate == 2) ? Wg : Wo;
  out[idx] = f2bf(W[row * K2P + k]);
}

__global__ void pack_b(const float* __restrict__ bf1, const float* __restrict__ bi1,
                       const float* __restrict__ bg1, const float* __restrict__ bo1,
                       const float* __restrict__ bf2, const float* __restrict__ bi2,
                       const float* __restrict__ bg2, const float* __restrict__ bo2,
                       float* __restrict__ b1p, float* __restrict__ b2p) {
  int idx = blockIdx.x * 256 + threadIdx.x;
  if (idx < 1024) {
    int gate = idx >> 8, row = idx & 255;
    const float* b = (gate == 0) ? bf1 : (gate == 1) ? bi1 : (gate == 2) ? bg1 : bo1;
    b1p[idx] = b[row];
  } else if (idx < 1536) {
    int n = idx - 1024;
    int gate = n >> 7, row = n & 127;
    const float* b = (gate == 0) ? bf2 : (gate == 1) ? bi2 : (gate == 2) ? bg2 : bo2;
    b2p[n] = b[row];
  }
}

// ===================== persistent per-batch-tile LSTM =====================
__global__ __launch_bounds__(256) void lstm_main(
    const float* __restrict__ x,
    const unsigned short* __restrict__ w1p, const unsigned short* __restrict__ w2p,
    const float* __restrict__ b1p, const float* __restrict__ b2p,
    const float* __restrict__ g1, const float* __restrict__ be1,
    const float* __restrict__ g2, const float* __restrict__ be2,
    float* __restrict__ out) {
  __shared__ unsigned short combA1[16 * CA_STRIDE];  // [x_t | h1norm | pad] bf16
  __shared__ unsigned short combA2[16 * CA_STRIDE];  // [h1norm | h2norm]   bf16
  __shared__ float xstage[2][16 * XSTRIDE];          // async double-buffered x_t
  __shared__ float h1buf[16 * H1];
  __shared__ float h2buf[16 * H2];
  __shared__ float lng1[H1], lnb1[H1], lng2[H2], lnb2[H2];

  const int tid  = threadIdx.x;
  const int wave = tid >> 5;
  const int lane = tid & 31;
  const int lm   = lane & 15;   // N index inside a 16-tile (B/C/D layout)
  const int hi   = lane >> 4;
  const int b0   = blockIdx.x * 16;

  for (int i = tid; i < 16 * CA_STRIDE; i += 256) { combA1[i] = 0; combA2[i] = 0; }
  for (int i = tid; i < H1; i += 256) { lng1[i] = g1[i]; lnb1[i] = be1[i]; }
  for (int i = tid; i < H2; i += 256) { lng2[i] = g2[i]; lnb2[i] = be2[i]; }

  // kick off async stage of x_0 into buffer 0
  for (int i = tid; i < 16 * IN_F; i += 256) {
    int row = i / IN_F, f = i - row * IN_F;
    const float* g = x + ((size_t)(b0 + row) * SEQ + 0) * IN_F + f;
    float* l = &xstage[0][row * XSTRIDE + f];
#if USE_ASYNC
    async_cp_b32(g, l);
#else
    *l = *g;
#endif
  }

  // Cell state lives in VGPRs for all 512 steps.
  v8f c1s[2], c2s;
  c1s[0] = (v8f)(0.0f); c1s[1] = (v8f)(0.0f); c2s = (v8f)(0.0f);

  // Hoist bias broadcasts (constant over time).
  float bia1[2][4], bia2[4];
#pragma unroll
  for (int q = 0; q < 2; ++q) {
    const int jt = wave * 2 + q;
#pragma unroll
    for (int g = 0; g < 4; ++g) bia1[q][g] = b1p[(g * 16 + jt) * 16 + lm];
  }
#pragma unroll
  for (int g = 0; g < 4; ++g) bia2[g] = b2p[(g * 8 + wave) * 16 + lm];

  __syncthreads();

  for (int t = 0; t < SEQ; ++t) {
    const int buf = t & 1;

    // ---- phase 1: wait for staged x_t, convert to bf16, prefetch x_{t+1} ----
#if USE_ASYNC
    __builtin_amdgcn_s_wait_asynccnt(0);
#endif
    __syncthreads();   // all waves' async copies of x_t visible in LDS
    {
      const int row = tid >> 4, f0 = tid & 15;
      const float* xr = &xstage[buf][row * XSTRIDE];
      for (int f = f0; f < IN_F; f += 16)
        combA1[row * CA_STRIDE + f] = f2bf(xr[f]);
    }
    if (t + 1 < SEQ) {   // overlap next step's x DMA with this step's compute
      for (int i = tid; i < 16 * IN_F; i += 256) {
        int row = i / IN_F, f = i - row * IN_F;
        const float* g = x + ((size_t)(b0 + row) * SEQ + (t + 1)) * IN_F + f;
        float* l = &xstage[buf ^ 1][row * XSTRIDE + f];
#if USE_ASYNC
        async_cp_b32(g, l);
#else
        *l = *g;
#endif
      }
    }
    __syncthreads();

    // ---- phase 2: layer-1 GEMM; per-q pass keeps 4 live accumulators ----
    // (merging both q passes spills B-frags to scratch; verified in round 2)
#pragma unroll
    for (int q = 0; q < 2; ++q) {
      const int jt = wave * 2 + q;
      v8f accf = (v8f)(bia1[q][0]);
      v8f acci = (v8f)(bia1[q][1]);
      v8f accg = (v8f)(bia1[q][2]);
      v8f acco = (v8f)(bia1[q][3]);
      for (int kt = 0; kt < KT; ++kt) {
        v16bf a = load_a_frag(combA1, lm, kt, hi);
        v16bf bF = load_b_frag(w1p, (0 * 16 + jt) * KT + kt, lane);
        v16bf bI = load_b_frag(w1p, (1 * 16 + jt) * KT + kt, lane);
        v16bf bG = load_b_frag(w1p, (2 * 16 + jt) * KT + kt, lane);
        v16bf bO = load_b_frag(w1p, (3 * 16 + jt) * KT + kt, lane);
        accf = __builtin_amdgcn_wmma_f32_16x16x32_bf16(false, a, false, bF, (short)0, accf, false, false);
        acci = __builtin_amdgcn_wmma_f32_16x16x32_bf16(false, a, false, bI, (short)0, acci, false, false);
        accg = __builtin_amdgcn_wmma_f32_16x16x32_bf16(false, a, false, bG, (short)0, accg, false, false);
        acco = __builtin_amdgcn_wmma_f32_16x16x32_bf16(false, a, false, bO, (short)0, acco, false, false);
      }
      v8f c = c1s[q];
#pragma unroll
      for (int r = 0; r < 8; ++r) {
        float fg = sigm(accf[r]);
        float ig = sigm(acci[r]);
        float gg = tanh_fast(accg[r]);
        float og = sigm(acco[r]);
        float cn = fg * c[r] + ig * gg;
        c[r] = cn;
        const int m = r + 8 * hi;               // C/D layout: M = r + 8*hi
        h1buf[m * H1 + jt * 16 + lm] = og * tanh_fast(cn);
      }
      c1s[q] = c;
    }
    __syncthreads();

    // ---- phase 3: LayerNorm(256); 2 rows/wave, 16-lane shuffle reduce ----
    {
      const int row = wave * 2 + hi;
      const float* hb = h1buf + row * H1;
      const int f0 = lm * 16;
      float v[16], s = 0.0f, s2 = 0.0f;
#pragma unroll
      for (int i = 0; i < 16; ++i) { float a = hb[f0 + i]; v[i] = a; s += a; s2 += a * a; }
#pragma unroll
      for (int m = 1; m < 16; m <<= 1) { s += __shfl_xor(s, m, 32); s2 += __shfl_xor(s2, m, 32); }
      const float mu  = s * (1.0f / H1);
      const float var = s2 * (1.0f / H1) - mu * mu;
      const float inv = rsqrtf(var + EPSL);
#pragma unroll
      for (int i = 0; i < 16; ++i) {
        const int f = f0 + i;
        const unsigned short hb16 = f2bf((v[i] - mu) * inv * lng1[f] + lnb1[f]);
        combA1[row * CA_STRIDE + IN_F + f] = hb16;   // recurrent input, next step
        combA2[row * CA_STRIDE + f]        = hb16;   // layer-2 input, this step
      }
    }
    __syncthreads();

    // ---- phase 4: layer-2 GEMM (1 feature tile per wave) ----
    {
      const int jt = wave;
      v8f accf = (v8f)(bia2[0]);
      v8f acci = (v8f)(bia2[1]);
      v8f accg = (v8f)(bia2[2]);
      v8f acco = (v8f)(bia2[3]);
      for (int kt = 0; kt < KT; ++kt) {
        v16bf a = load_a_frag(combA2, lm, kt, hi);
        v16bf bF = load_b_frag(w2p, (0 * 8 + jt) * KT + kt, lane);
        v16bf bI = load_b_frag(w2p, (1 * 8 + jt) * KT + kt, lane);
        v16bf bG = load_b_frag(w2p, (2 * 8 + jt) * KT + kt, lane);
        v16bf bO = load_b_frag(w2p, (3 * 8 + jt) * KT + kt, lane);
        accf = __builtin_amdgcn_wmma_f32_16x16x32_bf16(false, a, false, bF, (short)0, accf, false, false);
        acci = __builtin_amdgcn_wmma_f32_16x16x32_bf16(false, a, false, bI, (short)0, acci, false, false);
        accg = __builtin_amdgcn_wmma_f32_16x16x32_bf16(false, a, false, bG, (short)0, accg, false, false);
        acco = __builtin_amdgcn_wmma_f32_16x16x32_bf16(false, a, false, bO, (short)0, acco, false, false);
      }
#pragma unroll
      for (int r = 0; r < 8; ++r) {
        float fg = sigm(accf[r]);
        float ig = sigm(acci[r]);
        float gg = tanh_fast(accg[r]);
        float og = sigm(acco[r]);
        float cn = fg * c2s[r] + ig * gg;
        c2s[r] = cn;
        const int m = r + 8 * hi;
        h2buf[m * H2 + jt * 16 + lm] = og * tanh_fast(cn);
      }
    }
    __syncthreads();

    // ---- phase 5: LayerNorm(128); write recurrent input + final output ----
    {
      const int row = wave * 2 + hi;
      const float* hb = h2buf + row * H2;
      const int f0 = lm * 8;
      float v[8], s = 0.0f, s2 = 0.0f;
#pragma unroll
      for (int i = 0; i < 8; ++i) { float a = hb[f0 + i]; v[i] = a; s += a; s2 += a * a; }
#pragma unroll
      for (int m = 1; m < 16; m <<= 1) { s += __shfl_xor(s, m, 32); s2 += __shfl_xor(s2, m, 32); }
      const float mu  = s * (1.0f / H2);
      const float var = s2 * (1.0f / H2) - mu * mu;
      const float inv = rsqrtf(var + EPSL);
#pragma unroll
      for (int i = 0; i < 8; ++i) {
        const int f = f0 + i;
        const float hn = (v[i] - mu) * inv * lng2[f] + lnb2[f];
        combA2[row * CA_STRIDE + H1 + f] = f2bf(hn);
        if (t == SEQ - 1) out[(size_t)(b0 + row) * H2 + f] = hn;
      }
    }
    __syncthreads();
  }
}

extern "C" void kernel_launch(void* const* d_in, const int* in_sizes, int n_in,
                              void* d_out, int out_size, void* d_ws, size_t ws_size,
                              hipStream_t stream) {
  const float* x   = (const float*)d_in[0];
  const float* Wf1 = (const float*)d_in[1];
  const float* Wi1 = (const float*)d_in[2];
  const float* Wg1 = (const float*)d_in[3];
  const float* Wo1 = (const float*)d_in[4];
  const float* bf1 = (const float*)d_in[5];
  const float* bi1 = (const float*)d_in[6];
  const float* bg1 = (const float*)d_in[7];
  const float* bo1 = (const float*)d_in[8];
  const float* Wf2 = (const float*)d_in[9];
  const float* Wi2 = (const float*)d_in[10];
  const float* Wg2 = (const float*)d_in[11];
  const float* Wo2 = (const float*)d_in[12];
  const float* bf2 = (const float*)d_in[13];
  const float* bi2 = (const float*)d_in[14];
  const float* bg2 = (const float*)d_in[15];
  const float* bo2 = (const float*)d_in[16];
  const float* g1  = (const float*)d_in[17];
  const float* be1 = (const float*)d_in[18];
  const float* g2  = (const float*)d_in[19];
  const float* be2 = (const float*)d_in[20];
  float* out = (float*)d_out;

  unsigned short* w1p = (unsigned short*)d_ws;                 // 64*12*512 halfs
  unsigned short* w2p = w1p + (size_t)NT1 * KT * 512;          // 32*12*512 halfs
  float* b1p = (float*)(w2p + (size_t)NT2 * KT * 512);         // 1024 f32
  float* b2p = b1p + 1024;                                     // 512 f32

  pack_w1<<<(NT1 * KT * 512 + 255) / 256, 256, 0, stream>>>(Wf1, Wi1, Wg1, Wo1, w1p);
  pack_w2<<<(NT2 * KT * 512 + 255) / 256, 256, 0, stream>>>(Wf2, Wi2, Wg2, Wo2, w2p);
  pack_b<<<6, 256, 0, stream>>>(bf1, bi1, bg1, bo1, bf2, bi2, bg2, bo2, b1p, b2p);

  lstm_main<<<32, 256, 0, stream>>>(x, w1p, w2p, b1p, b2p, g1, be1, g2, be2, out);
}
